// VectorQuantizer_85487029059590
// MI455X (gfx1250) — matrix operand order repaired and checked by reading, not compiled, not run on hardware
//
#include <hip/hip_runtime.h>

// ---------- CDNA5 WMMA vector types ----------
typedef __attribute__((ext_vector_type(16))) _Float16 v16h;
typedef __attribute__((ext_vector_type(8)))  float    v8f;
typedef __attribute__((ext_vector_type(4)))  _Float16 h4;

#define N_CODES 1024
#define D       64
#define ROWS_TOTAL      524288            // 33554432 / 64
#define WAVES_PER_BLOCK 8
#define TILES_PER_WAVE  4                 // 4 * 16 = 64 rows per wave
#define ROWS_PER_WAVE   (TILES_PER_WAVE * 16)
#define ROWS_PER_BLOCK  (WAVES_PER_BLOCK * ROWS_PER_WAVE)   // 512
#define NBLOCKS         (ROWS_TOTAL / ROWS_PER_BLOCK)       // 1024
#define NPARTIALS       (NBLOCKS * WAVES_PER_BLOCK)         // 8192
#define Z_ELEMS         (ROWS_TOTAL * D)                    // 33554432

// =====================================================================
// Kernel 1: codebook fp32 -> f16, plus ||e||^2 per code.
// =====================================================================
__global__ void vq_prep(const float* __restrict__ cb,
                        _Float16* __restrict__ cbh,
                        float* __restrict__ norms) {
    int code = blockIdx.x * blockDim.x + threadIdx.x;
    if (code >= N_CODES) return;
    const float4* src = (const float4*)(cb + code * D);
    float s = 0.0f;
#pragma unroll
    for (int j = 0; j < D / 4; ++j) {
        float4 v = src[j];
        s += v.x * v.x + v.y * v.y + v.z * v.z + v.w * v.w;
        h4 h;
        h[0] = (_Float16)v.x; h[1] = (_Float16)v.y;
        h[2] = (_Float16)v.z; h[3] = (_Float16)v.w;
        *(h4*)(cbh + code * D + 4 * j) = h;
    }
    norms[code] = s;
}

// Build one A K-chunk fragment: lanes 0-15 (this-lane M row) need K in
// [off, off+8) and [off+8, off+16)  (off = 16 for lanes 16-31),
// pre-scaled by 2 so the WMMA directly yields 2*z.e.
__device__ __forceinline__ v16h make_a_frag(const float* __restrict__ p0,
                                            const float* __restrict__ p1) {
    v16h r;
#pragma unroll
    for (int i = 0; i < 8; ++i) r[i]     = (_Float16)(2.0f * p0[i]);
#pragma unroll
    for (int i = 0; i < 8; ++i) r[8 + i] = (_Float16)(2.0f * p1[i]);
    return r;
}

// =====================================================================
// Kernel 2: argmax_n (2 z.e_n - ||e_n||^2) via v_wmma_f32_16x16x32_f16,
// then gather z_q + per-wave loss partial.
// =====================================================================
__global__ __launch_bounds__(256) void vq_main(
    const float* __restrict__ z, const float* __restrict__ cb,
    const _Float16* __restrict__ cbh, const float* __restrict__ norms,
    float* __restrict__ out, float* __restrict__ partials) {

    __shared__ int s_idx[WAVES_PER_BLOCK * ROWS_PER_WAVE];

    const int tid  = threadIdx.x;
    const int wave = tid >> 5;
    const int lane = tid & 31;
    const int lhi  = lane >> 4;     // 0: lanes 0-15, 1: lanes 16-31
    const int llo  = lane & 15;
    const long wr0 = (long)blockIdx.x * ROWS_PER_BLOCK + (long)wave * ROWS_PER_WAVE;

    // ---- A fragments: 4 row-tiles x 2 K-chunks, converted once, live in VGPRs
    v16h a[TILES_PER_WAVE][2];
#pragma unroll
    for (int t = 0; t < TILES_PER_WAVE; ++t) {
        const float* zr  = z + (wr0 + t * 16 + llo) * D;
        const int    off = lhi * 16;
#pragma unroll
        for (int c = 0; c < 2; ++c)
            a[t][c] = make_a_frag(zr + c * 32 + off, zr + c * 32 + off + 8);
    }

    // ---- running best (score, index) per C/D row this lane owns
    float bs[TILES_PER_WAVE][8];
    int   bi[TILES_PER_WAVE][8];
#pragma unroll
    for (int t = 0; t < TILES_PER_WAVE; ++t)
#pragma unroll
        for (int v = 0; v < 8; ++v) { bs[t][v] = -3.4e38f; bi[t][v] = 0; }

    // ---- sweep the codebook, 16 codes per WMMA column tile
    for (int nb = 0; nb < N_CODES; nb += 16) {
        const int   n   = nb + llo;           // this lane's code column N
        const float nrm = norms[n];
        // B 32x16 f16 layout: lane holds col N=llo, K = 16*lhi .. +16 contiguous
        const _Float16* bp = cbh + (long)n * D + lhi * 16;
        const v16h b0 = *(const v16h*)(bp);        // K-chunk 0 (K 0..31)
        const v16h b1 = *(const v16h*)(bp + 32);   // K-chunk 1 (K 32..63)

#pragma unroll
        for (int t = 0; t < TILES_PER_WAVE; ++t) {
            v8f acc;
#pragma unroll
            for (int v = 0; v < 8; ++v) acc[v] = -nrm;  // exact fp32 -||e||^2 bias
            acc = __builtin_amdgcn_wmma_f32_16x16x32_f16(
                false, a[t][0], false, b0, (short)0, acc, false, false);
            acc = __builtin_amdgcn_wmma_f32_16x16x32_f16(
                false, a[t][1], false, b1, (short)0, acc, false, false);
#pragma unroll
            for (int v = 0; v < 8; ++v) {
                if (acc[v] > bs[t][v]) { bs[t][v] = acc[v]; bi[t][v] = n; }
            }
        }
    }

    // ---- reduce argmax across the 16 lanes sharing each M row
    //      (wave32: xor masks 1,2,4,8 stay inside each 16-lane group)
#pragma unroll
    for (int t = 0; t < TILES_PER_WAVE; ++t) {
#pragma unroll
        for (int v = 0; v < 8; ++v) {
            float s = bs[t][v];
            int   i = bi[t][v];
#pragma unroll
            for (int m = 1; m <= 8; m <<= 1) {
                float os = __shfl_xor(s, m, 32);
                int   oi = __shfl_xor(i, m, 32);
                if (os > s || (os == s && oi < i)) { s = os; i = oi; }
            }
            // lane with llo==v (in each half-group) publishes row M = 8*lhi + v
            if (llo == v) s_idx[wave * ROWS_PER_WAVE + t * 16 + lhi * 8 + v] = i;
        }
    }
    __syncthreads();

    // ---- gather z_q rows (fp32 codebook), write out, accumulate loss
    float lsum = 0.0f;
    const int col = llo * 4;                  // 16 lanes x float4 = one 64-el row
    for (int r = 0; r < ROWS_PER_WAVE; r += 2) {
        const int  row = r + lhi;             // 2 rows per iteration, coalesced
        const int  idx = s_idx[wave * ROWS_PER_WAVE + row];
        const long g   = (wr0 + row) * D + col;
        float4 zv = *(const float4*)(z  + g);
        float4 cv = *(const float4*)(cb + (long)idx * D + col);
        *(float4*)(out + g) = cv;
        float dx = cv.x - zv.x, dy = cv.y - zv.y;
        float dz = cv.z - zv.z, dw = cv.w - zv.w;
        lsum += dx * dx + dy * dy + dz * dz + dw * dw;
    }
#pragma unroll
    for (int m = 1; m < 32; m <<= 1) lsum += __shfl_xor(lsum, m, 32);
    if (lane == 0)
        partials[blockIdx.x * WAVES_PER_BLOCK + wave] =
            lsum * (1.25f / (float)Z_ELEMS);   // (1 + commitment) * mean
}

// =====================================================================
// Kernel 3: deterministic fixed-order reduction of loss partials.
// =====================================================================
__global__ void vq_loss(const float* __restrict__ partials,
                        float* __restrict__ out_loss) {
    __shared__ float sm[256];
    const int t = threadIdx.x;
    float s = 0.0f;
    for (int i = t; i < NPARTIALS; i += 256) s += partials[i];
    sm[t] = s;
    __syncthreads();
    for (int off = 128; off > 0; off >>= 1) {
        if (t < off) sm[t] += sm[t + off];
        __syncthreads();
    }
    if (t == 0) *out_loss = sm[0];
}

// =====================================================================
extern "C" void kernel_launch(void* const* d_in, const int* in_sizes, int n_in,
                              void* d_out, int out_size, void* d_ws, size_t ws_size,
                              hipStream_t stream) {
    const float* z  = (const float*)d_in[0];   // [524288,64]  fp32
    const float* cb = (const float*)d_in[1];   // [1024,64]    fp32
    float* out = (float*)d_out;                // z_q (33554432) ++ loss (1)

    // workspace layout
    _Float16* cbh      = (_Float16*)d_ws;                                  // 128 KB
    float*    norms    = (float*)((char*)d_ws + N_CODES * D * 2);          //   4 KB
    float*    partials = (float*)((char*)d_ws + N_CODES * D * 2 + N_CODES * 4); // 32 KB

    vq_prep<<<(N_CODES + 255) / 256, 256, 0, stream>>>(cb, cbh, norms);
    vq_main<<<NBLOCKS, 256, 0, stream>>>(z, cb, cbh, norms, out, partials);
    vq_loss<<<1, 256, 0, stream>>>(partials, out + Z_ELEMS);
}
